// SelfAttention2_82420422410272
// MI455X (gfx1250) — matrix operand order repaired and checked by reading, not compiled, not run on hardware
//
#include <hip/hip_runtime.h>
#include <hip/hip_bf16.h>
#include <cmath>
#include <cstdint>

// Shapes from the reference
#define B_   8
#define S_   2048
#define D_   1024
#define EPS_ 1e-7f

typedef __attribute__((ext_vector_type(16))) __bf16 v16bf;
typedef __attribute__((ext_vector_type(8)))  float  v8f;
typedef __attribute__((ext_vector_type(2)))  __bf16 bf16x2;

typedef unsigned int u32x4 __attribute__((ext_vector_type(4)));
typedef int          i32x4 __attribute__((ext_vector_type(4)));
typedef int          i32x8 __attribute__((ext_vector_type(8)));

// ---------------------------------------------------------------------------
// WMMA fragment loaders (16-bit 16x32 A / 32x16 B, this code's contiguous k order k = 16*(lane/16) + s).
// Memory must be "k-contiguous": element (row/col, k) at base + rc*ld + k.
// ---------------------------------------------------------------------------
__device__ __forceinline__ v16bf frag_a_kcontig(const __bf16* base, int ld, int lane) {
    const int m  = lane & 15;
    const int hi = lane >> 4;                 // lanes 16-31 hold K=16..31
    const __bf16* row = base + (size_t)m * ld + hi * 16;
    v16bf a;
#pragma unroll
    for (int v = 0; v < 8; ++v) {
        const int k = 2 * v;
        bf16x2 p = *reinterpret_cast<const bf16x2*>(row + k);
        a[2 * v]     = p[0];
        a[2 * v + 1] = p[1];
    }
    return a;
}

__device__ __forceinline__ v16bf frag_b_kcontig(const __bf16* base, int ld, int lane) {
    const int n  = lane & 15;
    const int hi = lane >> 4;                 // lanes 16-31 hold K=16..31
    const __bf16* col = base + (size_t)n * ld + hi * 16;
    v16bf b;
#pragma unroll
    for (int v = 0; v < 8; ++v) {
        bf16x2 p = *reinterpret_cast<const bf16x2*>(col + 2 * v);
        b[2 * v]     = p[0];
        b[2 * v + 1] = p[1];
    }
    return b;
}

__device__ __forceinline__ v8f wmma_bf16(v16bf a, v16bf b, v8f c) {
    return __builtin_amdgcn_wmma_f32_16x16x32_bf16(false, a, false, b,
                                                   (short)0, c, false, false);
}

// ---------------------------------------------------------------------------
// Prep kernels: bf16 convert + transposes so every WMMA B-frag is k-contiguous
// ---------------------------------------------------------------------------
__global__ void k_cvt_x(const float* __restrict__ x, __bf16* __restrict__ xb, int n) {
    int i = blockIdx.x * blockDim.x + threadIdx.x;
    if (i < n) xb[i] = (__bf16)x[i];
}

__global__ void k_transpose_w(const float* __restrict__ W, __bf16* __restrict__ Wt) {
    int i = blockIdx.x * blockDim.x + threadIdx.x;   // i = k*D + d
    int d = i & (D_ - 1);
    int k = i >> 10;
    Wt[(size_t)d * D_ + k] = (__bf16)W[i];
}

__global__ void k_transpose_x(const float* __restrict__ x, __bf16* __restrict__ xT) {
    int i = blockIdx.x * blockDim.x + threadIdx.x;   // i = (bb*S + t)*D + d
    int d   = i & (D_ - 1);
    int rem = i >> 10;                               // bb*S + t
    int t   = rem & (S_ - 1);
    int bb  = rem >> 11;
    xT[((size_t)bb * D_ + d) * S_ + t] = (__bf16)x[i];
}

// ---------------------------------------------------------------------------
// xt = x @ W + b.  4-wide register tiling + explicit double-buffered
// fragment pipeline: next K-step's loads are issued before this step's WMMAs.
// ---------------------------------------------------------------------------
__global__ void __launch_bounds__(128)
k_xw(const __bf16* __restrict__ xb, const __bf16* __restrict__ Wt,
     const float* __restrict__ bias, __bf16* __restrict__ xtb) {
    const int lane = threadIdx.x & 31;
    const int wave = threadIdx.x >> 5;
    const int m0    = blockIdx.x * 16;                       // row tile of B*S
    const int nbase = (blockIdx.y * 16 + wave * 4) * 16;     // 4 col tiles of D

    const __bf16* abase = xb + (size_t)m0 * D_;
    const __bf16* bbase[4];
#pragma unroll
    for (int j = 0; j < 4; ++j) bbase[j] = Wt + (size_t)(nbase + j * 16) * D_;

    v8f acc[4] = {};
    v16bf a_c = frag_a_kcontig(abase, D_, lane);
    v16bf b_c[4];
#pragma unroll
    for (int j = 0; j < 4; ++j) b_c[j] = frag_b_kcontig(bbase[j], D_, lane);

#pragma unroll 2
    for (int kk = 0; kk < D_ - 32; kk += 32) {
        const int kn = kk + 32;
        v16bf a_n = frag_a_kcontig(abase + kn, D_, lane);
        v16bf b_n[4];
#pragma unroll
        for (int j = 0; j < 4; ++j) b_n[j] = frag_b_kcontig(bbase[j] + kn, D_, lane);
#pragma unroll
        for (int j = 0; j < 4; ++j) acc[j] = wmma_bf16(a_c, b_c[j], acc[j]);
        a_c = a_n;
#pragma unroll
        for (int j = 0; j < 4; ++j) b_c[j] = b_n[j];
    }
#pragma unroll
    for (int j = 0; j < 4; ++j) acc[j] = wmma_bf16(a_c, b_c[j], acc[j]);

    const int hi = lane >> 4;
    const int n  = lane & 15;
#pragma unroll
    for (int j = 0; j < 4; ++j) {
        const int n0 = nbase + j * 16;
        const float bv = bias[n0 + n];
#pragma unroll
        for (int r = 0; r < 8; ++r) {
            const int row = m0 + r + 8 * hi;                 // C-layout: M = r + 8*hi
            xtb[(size_t)row * D_ + n0 + n] = (__bf16)(acc[j][r] + bv);
        }
    }
}

// ---------------------------------------------------------------------------
// Fused: scores = xt @ x^T, elementwise exp(tanh(w*lw))*offdiag, rowsum,
//        out = (w / (rowsum+eps)) @ x.   One 16-row block per workgroup;
//        the score panel lives only in LDS (never hits HBM).  The xt row
//        panel is staged into LDS by the Tensor Data Mover (contiguous 32KB).
// ---------------------------------------------------------------------------
__global__ void __launch_bounds__(256)
k_attn(const __bf16* __restrict__ xb,    // [B*S, D]  bf16, row-major
       const __bf16* __restrict__ xTb,   // [B, D, S] bf16 (x transposed per batch)
       const __bf16* __restrict__ xtb,   // [B*S, D]  bf16 (x@W+b)
       float* __restrict__ out) {        // [B, S, D] f32
    __shared__ __align__(16) __bf16 aT[16 * D_];   // 32 KB: xt row panel
    __shared__ __align__(16) __bf16 wR[16 * S_];   // 64 KB: score row panel
    __shared__ float rsum[16];

    const int tid  = threadIdx.x;
    const int lane = tid & 31;
    const int wave = tid >> 5;
    const int bb = blockIdx.x >> 7;               // 128 row-tiles per batch
    const int m0 = (blockIdx.x & 127) * 16;

    // ---- TDM: DMA the contiguous 16x1024 bf16 xt panel into LDS ----
    if (wave == 0) {
        const uint64_t gaddr =
            (uint64_t)(uintptr_t)(xtb + (size_t)(bb * S_ + m0) * D_);
        const uint32_t lds = (uint32_t)(uintptr_t)aT;

        u32x4 g0;
        g0[0] = 1u;                                        // count=1 (valid D#)
        g0[1] = lds;                                       // lds_addr
        g0[2] = (uint32_t)gaddr;                           // global_addr[31:0]
        g0[3] = (uint32_t)((gaddr >> 32) & 0x1FFFFFFu)     // global_addr[56:32]
              | 0x80000000u;                               // type=2 ("image")

        i32x8 g1;
        g1[0] = 0x00010000;                 // workgroup_mask=0, data_size=1 (2B)
        g1[1] = (int)(16384u << 16);        // tensor_dim0[15:0] = 16384 elems
        g1[2] = (int)(1u << 16);            // tensor_dim0[31:16]=0, tensor_dim1=1
        g1[3] = (int)(16384u << 16);        // tensor_dim1 hi=0, tile_dim0=16384
        g1[4] = 0;                          // tile_dim1=0 (unused), tile_dim2=0
        g1[5] = 16384;                      // tensor_dim0_stride[31:0]
        g1[6] = 0;                          // stride hi, tensor_dim1_stride lo
        g1[7] = 0;
        i32x4 g2 = {0, 0, 0, 0};
        i32x4 g3 = {0, 0, 0, 0};
#if __clang_major__ >= 23
        i32x8 g4 = {0, 0, 0, 0, 0, 0, 0, 0};
        __builtin_amdgcn_tensor_load_to_lds(g0, g1, g2, g3, g4, 0);
#else
        __builtin_amdgcn_tensor_load_to_lds(g0, g1, g2, g3, 0);
#endif
        __builtin_amdgcn_s_wait_tensorcnt(0);
    }
    if (tid < 16) rsum[tid] = 0.0f;
    __syncthreads();

    const int hi = lane >> 4;
    const int n  = lane & 15;

    // ---- score phase: 16 x 2048 panel; each wave owns 16 t-tiles,
    //      4 at a time, with a double-buffered fragment pipeline ----
    float psum[8];
#pragma unroll
    for (int r = 0; r < 8; ++r) psum[r] = 0.0f;

    for (int g = 0; g < 4; ++g) {
        const __bf16* bbase[4];
#pragma unroll
        for (int j = 0; j < 4; ++j) {
            const int t0 = (wave + 8 * (4 * g + j)) * 16;
            bbase[j] = xb + (size_t)(bb * S_ + t0) * D_;
        }
        v8f acc[4] = {};
        v16bf a_c = frag_a_kcontig(aT, D_, lane);
        v16bf b_c[4];
#pragma unroll
        for (int j = 0; j < 4; ++j) b_c[j] = frag_b_kcontig(bbase[j], D_, lane);

#pragma unroll 2
        for (int kk = 0; kk < D_ - 32; kk += 32) {
            const int kn = kk + 32;
            v16bf a_n = frag_a_kcontig(aT + kn, D_, lane);
            v16bf b_n[4];
#pragma unroll
            for (int j = 0; j < 4; ++j) b_n[j] = frag_b_kcontig(bbase[j] + kn, D_, lane);
#pragma unroll
            for (int j = 0; j < 4; ++j) acc[j] = wmma_bf16(a_c, b_c[j], acc[j]);
            a_c = a_n;
#pragma unroll
            for (int j = 0; j < 4; ++j) b_c[j] = b_n[j];
        }
#pragma unroll
        for (int j = 0; j < 4; ++j) acc[j] = wmma_bf16(a_c, b_c[j], acc[j]);

#pragma unroll
        for (int j = 0; j < 4; ++j) {
            const int t0 = (wave + 8 * (4 * g + j)) * 16;
            const int tg = t0 + n;
#pragma unroll
            for (int r = 0; r < 8; ++r) {
                const int M  = r + 8 * hi;
                const int sg = m0 + M;
                const float loc = fabsf((float)(sg - tg));
                const float lw  = (sg == tg) ? 0.0f : (1.0f / (loc + EPS_));
                float v = __expf(tanhf(acc[j][r] * lw));
                if (sg == tg) v = 0.0f;               // off-diagonal mask
                wR[M * S_ + tg] = (__bf16)v;
                psum[r] += v;
            }
        }
    }
#pragma unroll
    for (int r = 0; r < 8; ++r) atomicAdd(&rsum[r + 8 * hi], psum[r]);
    __syncthreads();

    // ---- output phase: out = wR @ x, K dim = S; each wave owns 8 d-tiles,
    //      4 at a time, double-buffered ----
    for (int g = 0; g < 2; ++g) {
        const __bf16* bbase[4];
#pragma unroll
        for (int j = 0; j < 4; ++j) {
            const int d0 = (wave + 8 * (4 * g + j)) * 16;
            bbase[j] = xTb + ((size_t)bb * D_ + d0) * S_;
        }
        v8f acc[4] = {};
        v16bf a_c = frag_a_kcontig(wR, S_, lane);
        v16bf b_c[4];
#pragma unroll
        for (int j = 0; j < 4; ++j) b_c[j] = frag_b_kcontig(bbase[j], S_, lane);

#pragma unroll 2
        for (int tt2 = 0; tt2 < S_ - 32; tt2 += 32) {
            const int tn = tt2 + 32;
            v16bf a_n = frag_a_kcontig(wR + tn, S_, lane);
            v16bf b_n[4];
#pragma unroll
            for (int j = 0; j < 4; ++j) b_n[j] = frag_b_kcontig(bbase[j] + tn, S_, lane);
#pragma unroll
            for (int j = 0; j < 4; ++j) acc[j] = wmma_bf16(a_c, b_c[j], acc[j]);
            a_c = a_n;
#pragma unroll
            for (int j = 0; j < 4; ++j) b_c[j] = b_n[j];
        }
#pragma unroll
        for (int j = 0; j < 4; ++j) acc[j] = wmma_bf16(a_c, b_c[j], acc[j]);

#pragma unroll
        for (int j = 0; j < 4; ++j) {
            const int d0 = (wave + 8 * (4 * g + j)) * 16;
#pragma unroll
            for (int r = 0; r < 8; ++r) {
                const int M = r + 8 * hi;
                const float den = rsum[M] + EPS_;
                out[(size_t)(bb * S_ + m0 + M) * D_ + d0 + n] = acc[j][r] / den;
            }
        }
    }
}

// ---------------------------------------------------------------------------
extern "C" void kernel_launch(void* const* d_in, const int* in_sizes, int n_in,
                              void* d_out, int out_size, void* d_ws, size_t ws_size,
                              hipStream_t stream) {
    const float* x = (const float*)d_in[0];   // [B,S,D]
    const float* W = (const float*)d_in[1];   // [D,D]
    const float* b = (const float*)d_in[2];   // [D]
    float* out = (float*)d_out;

    // Workspace carve-up (bf16 staging buffers)
    const size_t XB_BYTES = (size_t)B_ * S_ * D_ * 2;   // 32 MB each
    char* ws = (char*)d_ws;
    __bf16* xb  = (__bf16*)(ws);                        // x as bf16    [B*S, D]
    __bf16* xTb = (__bf16*)(ws + XB_BYTES);             // x^T as bf16  [B, D, S]
    __bf16* xtb = (__bf16*)(ws + 2 * XB_BYTES);         // x@W+b bf16   [B*S, D]
    __bf16* Wt  = (__bf16*)(ws + 3 * XB_BYTES);         // W^T as bf16  [D, D]

    const int nElem = B_ * S_ * D_;
    k_cvt_x      <<<nElem / 256, 256, 0, stream>>>(x, xb, nElem);
    k_transpose_x<<<nElem / 256, 256, 0, stream>>>(x, xTb);
    k_transpose_w<<<(D_ * D_) / 256, 256, 0, stream>>>(W, Wt);

    dim3 g2(B_ * S_ / 16, D_ / 256);     // 1024 x 4 blocks, 4 waves x 4 N-tiles
    k_xw<<<g2, 128, 0, stream>>>(xb, Wt, b, xtb);

    k_attn<<<B_ * (S_ / 16), 256, 0, stream>>>(xb, xTb, xtb, out);
}
